// TransBlock_35107062678093
// MI455X (gfx1250) — matrix operand order repaired and checked by reading, not compiled, not run on hardware
//
#include <hip/hip_runtime.h>

// =====================================================================
// CDNA5 / gfx1250 fused transformer block (wave32, WMMA f16 16x16x32,
// async global->LDS double-buffered staging)
// =====================================================================

typedef _Float16 h16;
typedef __attribute__((ext_vector_type(16))) _Float16 v16h;
typedef __attribute__((ext_vector_type(8)))  float    v8f;
typedef __attribute__((ext_vector_type(4)))  int      i32x4;

#define LDSPITCH 40   // 32 K-elems + 8 pad (h16) -> 80B row pitch, 16B aligned

#define AS1 __attribute__((address_space(1)))
#define AS3 __attribute__((address_space(3)))

#if defined(__has_builtin)
#if __has_builtin(__builtin_amdgcn_global_load_async_to_lds_b128)
#define HAS_ASYNC_LDS 1
#endif
#endif
#ifndef HAS_ASYNC_LDS
#define HAS_ASYNC_LDS 0
#endif

// 16B global -> LDS copy. Async path uses the CDNA5
// GLOBAL_LOAD_ASYNC_TO_LDS_B128 (ASYNCcnt); fallback is a sync VGPR copy.
// Builtin signature (from compiler diagnostics): (int4 AS1*, int4 AS3*, Ii, Ii)
__device__ __forceinline__ void async_copy16(const h16* g, h16* l) {
#if HAS_ASYNC_LDS
  __builtin_amdgcn_global_load_async_to_lds_b128(
      (AS1 i32x4*)(unsigned long long)(const void*)g,
      (AS3 i32x4*)(unsigned int)(unsigned long long)(const void*)l,
      0, 0);
#else
  *reinterpret_cast<float4*>(l) = *reinterpret_cast<const float4*>(g);
#endif
}

__device__ __forceinline__ void wait_async() {
#if HAS_ASYNC_LDS
#if __has_builtin(__builtin_amdgcn_s_wait_asynccnt)
  __builtin_amdgcn_s_wait_asynccnt(0);
#else
  asm volatile("s_wait_asynccnt 0" ::: "memory");
#endif
#endif
}

// stage one 128x32 A-tile and 128x32 B-tile into LDS (4 x 16B per thread)
__device__ __forceinline__ void stage_tile(const h16* Ag, int lda,
                                           const h16* Bg, int ldb,
                                           h16* As, h16* Bs, int k0, int tid) {
#pragma unroll
  for (int c = tid; c < 512; c += 256) {
    const int row = c >> 2, col = (c & 3) << 3;
    async_copy16(Ag + (size_t)row * lda + k0 + col, As + row * LDSPITCH + col);
    async_copy16(Bg + (size_t)row * ldb + k0 + col, Bs + row * LDSPITCH + col);
  }
}

// ---- WMMA fragment loaders (layouts per CDNA5 ISA 7.12.2) -----------
__device__ __forceinline__ v16h frag_a(const h16* s, int mbase, int l) {
  const int m  = mbase + (l & 15);
  const int lh = l >> 4;
  const h16* p = s + m * LDSPITCH + lh * 8;
  union { v16h v; float4 f[2]; } u;
  u.f[0] = *reinterpret_cast<const float4*>(p);
  u.f[1] = *reinterpret_cast<const float4*>(p + 16);
  return u.v;
}
__device__ __forceinline__ v16h frag_b(const h16* s, int nbase, int l) {
  const int n  = nbase + (l & 15);
  const int lh = l >> 4;
  const h16* p = s + n * LDSPITCH + lh * 8;
  union { v16h v; float4 f[2]; } u;
  u.f[0] = *reinterpret_cast<const float4*>(p);
  u.f[1] = *reinterpret_cast<const float4*>(p + 16);
  return u.v;
}

// ---- weight transpose+convert: in f32 [K][N] -> out f16 [N][K] ------
__global__ __launch_bounds__(256) void transpose_f32_to_f16(
    const float* __restrict__ in, h16* __restrict__ out, int K, int N) {
  const unsigned idx = blockIdx.x * 256u + threadIdx.x;
  if (idx >= (unsigned)(N * K)) return;
  const int n = idx / K, k = idx % K;
  out[idx] = (h16)in[(size_t)k * N + n];
}

// ---- layernorm over 512, f32 in -> f16 out --------------------------
__global__ __launch_bounds__(256) void layernorm_f32_to_f16(
    const float* __restrict__ in, const float* __restrict__ g,
    const float* __restrict__ bb, h16* __restrict__ out) {
  const int row = blockIdx.x;
  const int t = threadIdx.x;
  const float2 v = reinterpret_cast<const float2*>(in + (size_t)row * 512)[t];
  float s1 = v.x + v.y;
  float s2 = v.x * v.x + v.y * v.y;
#pragma unroll
  for (int o = 16; o > 0; o >>= 1) { s1 += __shfl_xor(s1, o); s2 += __shfl_xor(s2, o); }
  __shared__ float a1[8], a2[8], stat[2];
  const int wid = t >> 5;
  if ((t & 31) == 0) { a1[wid] = s1; a2[wid] = s2; }
  __syncthreads();
  if (t == 0) {
    float S1 = 0.f, S2 = 0.f;
    for (int i = 0; i < 8; ++i) { S1 += a1[i]; S2 += a2[i]; }
    const float mu  = S1 * (1.f / 512.f);
    const float var = S2 * (1.f / 512.f) - mu * mu;
    stat[0] = mu; stat[1] = rsqrtf(var + 1e-5f);
  }
  __syncthreads();
  const float mu = stat[0], ri = stat[1];
  const int c = t * 2;
  union { float f; h16 e[2]; } u;
  u.e[0] = (h16)((v.x - mu) * ri * g[c]     + bb[c]);
  u.e[1] = (h16)((v.y - mu) * ri * g[c + 1] + bb[c + 1]);
  reinterpret_cast<float*>(out + (size_t)row * 512)[t] = u.f;
}

// ---- generic WMMA GEMM: C[M][N] = A[M][K](f16) x Bt[N][K](f16) ------
// block tile 128x128, 8 waves (4 in M x 2 in N), wave tile 32x64.
// Double-buffered async LDS staging: tile k+1 streams in while tile k
// is consumed by WMMA; one s_wait_asynccnt + barrier per K-step.
// mode 0: out f16 = acc+bias               (qkv)
// mode 1: out f32 = acc+bias+resid[m*N+n]  (W0 / W2)
// mode 2: out f16 = relu(acc+bias)         (W1)
__global__ __launch_bounds__(256) void gemm_f16(
    const h16* __restrict__ A, int lda,
    const h16* __restrict__ Bt,
    const float* __restrict__ bias,
    const float* __restrict__ resid,
    void* __restrict__ outp,
    int M, int N, int K, int mode) {
  __shared__ __align__(16) h16 As[2][128 * LDSPITCH];
  __shared__ __align__(16) h16 Bs[2][128 * LDSPITCH];
  const int tid = threadIdx.x;
  const int m0 = blockIdx.y * 128, n0 = blockIdx.x * 128;
  const int l = tid & 31, wid = tid >> 5;
  const int wm = wid & 3, wn = wid >> 2;
  const h16* Ag = A + (size_t)m0 * lda;
  const h16* Bg = Bt + (size_t)n0 * K;

  v8f acc[2][4] = {};

  stage_tile(Ag, lda, Bg, K, As[0], Bs[0], 0, tid);
  wait_async();
  __syncthreads();

  for (int k0 = 0; k0 < K; k0 += 32) {
    const int p = (k0 >> 5) & 1;
    if (k0 + 32 < K) {
      stage_tile(Ag, lda, Bg, K, As[p ^ 1], Bs[p ^ 1], k0 + 32, tid);
      if (k0 + 64 < K) {  // L2 prefetch two tiles ahead
        __builtin_prefetch(Ag + (size_t)(tid >> 1) * lda + k0 + 64, 0, 1);
        __builtin_prefetch(Bg + (size_t)(tid >> 1) * K + k0 + 64, 0, 1);
      }
    }
    const v16h a0 = frag_a(As[p], wm * 32, l);
    const v16h a1 = frag_a(As[p], wm * 32 + 16, l);
#pragma unroll
    for (int j = 0; j < 4; ++j) {
      const v16h fb = frag_b(Bs[p], wn * 64 + j * 16, l);
      acc[0][j] = __builtin_amdgcn_wmma_f32_16x16x32_f16(false, a0, false, fb,
                                                         (short)0, acc[0][j], false, false);
      acc[1][j] = __builtin_amdgcn_wmma_f32_16x16x32_f16(false, a1, false, fb,
                                                         (short)0, acc[1][j], false, false);
    }
    wait_async();
    __syncthreads();
  }

  // epilogue: C/D layout -> lane l: N=l&15, M=r+8*(l>>4) per VGPR r
  const int nn = l & 15, lh = l >> 4;
#pragma unroll
  for (int i = 0; i < 2; ++i) {
#pragma unroll
    for (int j = 0; j < 4; ++j) {
      const int gn = n0 + wn * 64 + j * 16 + nn;
      const float bv = bias[gn];
#pragma unroll
      for (int r = 0; r < 8; ++r) {
        const int gm = m0 + wm * 32 + i * 16 + 8 * lh + r;
        float v = acc[i][j][r] + bv;
        const size_t o = (size_t)gm * N + gn;
        if (mode == 1) {
          ((float*)outp)[o] = v + resid[o];
        } else {
          if (mode == 2) v = fmaxf(v, 0.f);
          ((h16*)outp)[o] = (h16)v;
        }
      }
    }
  }
}

// ---- repack q,k from raw-reshape layout to [b,h,n,512] --------------
// q[b,n,id,h] = qkvflat[b*4194304 + n*4096 + id*8 + h] (+16777216 for k)
__global__ __launch_bounds__(256) void repack_qk(
    const h16* __restrict__ qkv, h16* __restrict__ q_l, h16* __restrict__ k_l) {
  const unsigned idx = blockIdx.x * 256u + threadIdx.x;  // 2^22 threads
  const int ch = idx & 63;
  const int n  = (idx >> 6) & 1023;
  const int bh = (idx >> 16) & 31;
  const int wch = idx >> 21;
  const int b = bh >> 3, h = bh & 7;
  const h16* src = qkv + (size_t)wch * 16777216u + (size_t)b * 4194304u +
                   (size_t)n * 4096u + ch * 64 + h;
  union { float4 f; h16 e[8]; } u;
#pragma unroll
  for (int i = 0; i < 8; ++i) u.e[i] = src[i * 8];
  h16* dst = (wch ? k_l : q_l) + ((size_t)bh * 1024 + n) * 512 + ch * 8;
  *reinterpret_cast<float4*>(dst) = u.f;
}

// ---- score column-softmax stats + diagonal via WMMA ----------------
// per block: (b,h), 128 t-columns; loop n-blocks of 128, online softmax
// over n (axis=1 of softmax in the reference), grab score[t,t].
__global__ __launch_bounds__(256) void attn_diag_kernel(
    const h16* __restrict__ q_l, const h16* __restrict__ k_l,
    float* __restrict__ diag_out) {
  const int tb = blockIdx.x;          // t-block 0..7
  const int bh = blockIdx.y;          // 0..31
  const int b = bh >> 3, h = bh & 7;
  const h16* Q  = q_l + (size_t)bh * 1024 * 512;
  const h16* Kt = k_l + (size_t)bh * 1024 * 512;
  const int t0 = tb * 128;

  __shared__ __align__(16) h16 Qs[2][128 * LDSPITCH];
  __shared__ __align__(16) h16 Ks[2][128 * LDSPITCH];
  __shared__ float sm[128][4], ss[128][4], sdiag[128];

  const int tid = threadIdx.x;
  const int l = tid & 31, wid = tid >> 5;
  const int wm = wid & 3, wn = wid >> 2;
  const int nn = l & 15, lh = l >> 4;
  const float scale = 0.044194173824159216f;  // 512^-0.5

  float m_run[4] = {-1e30f, -1e30f, -1e30f, -1e30f};
  float s_run[4] = {0.f, 0.f, 0.f, 0.f};

  const h16* Kg = Kt + (size_t)t0 * 512;

  for (int nb = 0; nb < 8; ++nb) {
    const h16* Qg = Q + (size_t)nb * 128 * 512;
    v8f acc[2][4] = {};

    stage_tile(Qg, 512, Kg, 512, Qs[0], Ks[0], 0, tid);
    wait_async();
    __syncthreads();

    for (int k0 = 0; k0 < 512; k0 += 32) {
      const int p = (k0 >> 5) & 1;
      if (k0 + 32 < 512)
        stage_tile(Qg, 512, Kg, 512, Qs[p ^ 1], Ks[p ^ 1], k0 + 32, tid);
      const v16h a0 = frag_a(Qs[p], wm * 32, l);
      const v16h a1 = frag_a(Qs[p], wm * 32 + 16, l);
#pragma unroll
      for (int j = 0; j < 4; ++j) {
        const v16h fb = frag_b(Ks[p], wn * 64 + j * 16, l);
        acc[0][j] = __builtin_amdgcn_wmma_f32_16x16x32_f16(false, a0, false, fb,
                                                           (short)0, acc[0][j], false, false);
        acc[1][j] = __builtin_amdgcn_wmma_f32_16x16x32_f16(false, a1, false, fb,
                                                           (short)0, acc[1][j], false, false);
      }
      wait_async();
      __syncthreads();
    }

    // online per-column (t) softmax stats over the n dimension
#pragma unroll
    for (int i = 0; i < 2; ++i) {
      const int gn0 = nb * 128 + wm * 32 + i * 16;
#pragma unroll
      for (int j = 0; j < 4; ++j) {
        float tmax = -1e30f;
#pragma unroll
        for (int r = 0; r < 8; ++r) tmax = fmaxf(tmax, acc[i][j][r]);
        tmax = fmaxf(tmax, __shfl_xor(tmax, 16));  // fold lane halves (M rows)
        tmax *= scale;
        const float nm = fmaxf(m_run[j], tmax);
        float ts = 0.f;
#pragma unroll
        for (int r = 0; r < 8; ++r) ts += __expf(acc[i][j][r] * scale - nm);
        ts += __shfl_xor(ts, 16);
        s_run[j] = s_run[j] * __expf(m_run[j] - nm) + ts;
        m_run[j] = nm;
        // diagonal score[t,t]: element with M==N in this tile
        const int gt0 = t0 + wn * 64 + j * 16;
        if (gn0 == gt0 && ((nn < 8) == (lh == 0))) {
          const int rr = nn & 7;
          float dv = 0.f;
#pragma unroll
          for (int r = 0; r < 8; ++r) if (r == rr) dv = acc[i][j][r];
          sdiag[wn * 64 + j * 16 + nn] = dv * scale;
        }
      }
    }
  }
  // combine across the 4 n-waves, emit attn diag
  if (lh == 0) {
#pragma unroll
    for (int j = 0; j < 4; ++j) {
      const int col = wn * 64 + j * 16 + nn;
      sm[col][wm] = m_run[j];
      ss[col][wm] = s_run[j];
    }
  }
  __syncthreads();
  if (tid < 128) {
    float M = sm[tid][0];
    for (int w = 1; w < 4; ++w) M = fmaxf(M, sm[tid][w]);
    float S = 0.f;
    for (int w = 0; w < 4; ++w) S += ss[tid][w] * __expf(sm[tid][w] - M);
    const float attn = __expf(sdiag[tid] - M) / S;
    const int t = t0 + tid;
    diag_out[((size_t)(b * 1024 + t)) * 8 + h] = attn;
  }
}

// ---- av = diag * v, in place on the v region (rows=tokens, 4096 wide)
__global__ __launch_bounds__(256) void scale_v(
    h16* __restrict__ vbuf, const float* __restrict__ diag) {
  const unsigned idx = blockIdx.x * 256u + threadIdx.x;  // 2,097,152
  const int r  = idx >> 9;
  const int c8 = idx & 511;
  h16* p = vbuf + (size_t)r * 4096 + c8 * 8;
  union { float4 f; h16 e[8]; } u;
  u.f = *reinterpret_cast<const float4*>(p);
  const float* dg = diag + (size_t)r * 8;   // h = c & 7 == lane within chunk
#pragma unroll
  for (int i = 0; i < 8; ++i) u.e[i] = (h16)((float)u.e[i] * dg[i]);
  *reinterpret_cast<float4*>(p) = u.f;
}

// =====================================================================
extern "C" void kernel_launch(void* const* d_in, const int* in_sizes, int n_in,
                              void* d_out, int out_size, void* d_ws, size_t ws_size,
                              hipStream_t stream) {
  (void)in_sizes; (void)n_in; (void)out_size; (void)ws_size;
  const float* x     = (const float*)d_in[0];
  const float* ln1_g = (const float*)d_in[1];
  const float* ln1_b = (const float*)d_in[2];
  const float* ln2_g = (const float*)d_in[3];
  const float* ln2_b = (const float*)d_in[4];
  const float* Wqkv  = (const float*)d_in[5];
  const float* bqkv  = (const float*)d_in[6];
  const float* W0    = (const float*)d_in[7];
  const float* b0    = (const float*)d_in[8];
  const float* W1    = (const float*)d_in[9];
  const float* b1    = (const float*)d_in[10];
  const float* W2    = (const float*)d_in[11];
  const float* b2    = (const float*)d_in[12];
  float* out = (float*)d_out;

  char* ws = (char*)d_ws;
  size_t off = 0;
  auto carve = [&](size_t bytes) -> char* {
    char* p = ws + off;
    off += (bytes + 255) & ~size_t(255);
    return p;
  };
  h16*   Wqkv_t = (h16*)carve(12288ull * 512 * 2);
  h16*   W0_t   = (h16*)carve(512ull * 4096 * 2);
  h16*   W1_t   = (h16*)carve(2048ull * 512 * 2);
  h16*   W2_t   = (h16*)carve(512ull * 2048 * 2);
  h16*   ln1a   = (h16*)carve(4096ull * 512 * 2);
  h16*   qkv    = (h16*)carve(50331648ull * 2);   // [4096][12288] f16
  h16*   q_l    = (h16*)carve(16777216ull * 2);   // [32][1024][512]
  h16*   k_l    = (h16*)carve(16777216ull * 2);
  float* diag   = (float*)carve(4096ull * 8 * 4); // [token][h]
  float* res    = (float*)carve(4096ull * 512 * 4);
  h16*   ln2a   = (h16*)carve(4096ull * 512 * 2);
  h16*   hdn    = (h16*)carve(4096ull * 2048 * 2);

  // weights -> f16 [N][K]
  transpose_f32_to_f16<<<(12288 * 512 + 255) / 256, 256, 0, stream>>>(Wqkv, Wqkv_t, 512, 12288);
  transpose_f32_to_f16<<<(512 * 4096 + 255) / 256, 256, 0, stream>>>(W0, W0_t, 4096, 512);
  transpose_f32_to_f16<<<(2048 * 512 + 255) / 256, 256, 0, stream>>>(W1, W1_t, 512, 2048);
  transpose_f32_to_f16<<<(512 * 2048 + 255) / 256, 256, 0, stream>>>(W2, W2_t, 2048, 512);

  // ln1 -> qkv GEMM
  layernorm_f32_to_f16<<<4096, 256, 0, stream>>>(x, ln1_g, ln1_b, ln1a);
  gemm_f16<<<dim3(96, 32), 256, 0, stream>>>(ln1a, 512, Wqkv_t, bqkv, nullptr,
                                             qkv, 4096, 12288, 512, 0);
  // attention diag
  repack_qk<<<4194304 / 256, 256, 0, stream>>>(qkv, q_l, k_l);
  attn_diag_kernel<<<dim3(8, 32), 256, 0, stream>>>(q_l, k_l, diag);
  scale_v<<<2097152 / 256, 256, 0, stream>>>(qkv + 33554432ull, diag);

  // av @ W0 + b0 + x -> res (v region rows are contiguous token rows)
  gemm_f16<<<dim3(4, 32), 256, 0, stream>>>(qkv + 33554432ull, 4096, W0_t, b0, x,
                                            res, 4096, 512, 4096, 1);
  // MLP
  layernorm_f32_to_f16<<<4096, 256, 0, stream>>>(res, ln2_g, ln2_b, ln2a);
  gemm_f16<<<dim3(16, 32), 256, 0, stream>>>(ln2a, 512, W1_t, b1, nullptr,
                                             hdn, 4096, 2048, 512, 2);
  gemm_f16<<<dim3(4, 32), 256, 0, stream>>>(hdn, 2048, W2_t, b2, res,
                                            out, 4096, 512, 2048, 1);
}
